// AssociativeMemoryCell_60370060313076
// MI455X (gfx1250) — compile-verified, matrix operand order checked by reading, not run on hardware
//
#include <hip/hip_runtime.h>

typedef __attribute__((ext_vector_type(16))) __bf16 bf16x16;
typedef __attribute__((ext_vector_type(8)))  float  v8f;

#define TID ((int)threadIdx.x)

constexpr int Bn = 4;        // batch
constexpr int S  = 2048;     // seq
constexpr int D  = 2048;     // d_model
constexpr int H  = 8;        // heads
constexpr int DH = 64;       // per-head q dim (512/8)
constexpr int DT = 256;      // per-head out dim (2048/8)
constexpr int DK = 384;      // phi dim (2*NU*512/8)
constexpr int NM = 128;      // NUM_MEM
constexpr float EPS = 1e-5f;

__device__ __forceinline__ unsigned short f2bf(float f){
  union { float f; unsigned u; } v; v.f = f;
  unsigned u = v.u;
  u += 0x7fffu + ((u >> 16) & 1u);      // round-to-nearest-even
  return (unsigned short)(u >> 16);
}

__device__ __forceinline__ v8f wmma_bf16(bf16x16 a, bf16x16 b, v8f c){
  return __builtin_amdgcn_wmma_f32_16x16x32_bf16(false, a, false, b, (short)0, c, false, false);
}

// Fragment loader for 16-bit A(16x32) (and B when stored N-major):
// lane m = lane&15; kb = lane<16?0:8; two 16B chunks at kb and kb+16.
__device__ __forceinline__ bf16x16 load_frag(const unsigned short* p, int ld, int row, int kc){
  int lane = TID & 31;
  const unsigned short* q = p + (size_t)(row + (lane & 15)) * ld + kc + ((lane < 16) ? 0 : 8);
  union { uint4 u[2]; bf16x16 v; } t;
  t.u[0] = *(const uint4*)q;
  t.u[1] = *(const uint4*)(q + 16);
  return t.v;
}

__device__ __forceinline__ float phix(const float* qrow, int p){
  return (p < 64) ? fmaxf(qrow[p], 0.f) : fmaxf(-qrow[p - 64], 0.f);
}

__device__ __forceinline__ uint4 cvt8(const float* src){
  const float4* p = (const float4*)src;
  float4 a = p[0], b = p[1];
  union { unsigned short s[8]; uint4 u; } o;
  o.s[0]=f2bf(a.x); o.s[1]=f2bf(a.y); o.s[2]=f2bf(a.z); o.s[3]=f2bf(a.w);
  o.s[4]=f2bf(b.x); o.s[5]=f2bf(b.y); o.s[6]=f2bf(b.z); o.s[7]=f2bf(b.w);
  return o.u;
}

// ---------------- elementwise fp32 -> bf16 pack (8 elems / thread) ----------------
__global__ __launch_bounds__(256) void k_cvt(const float* __restrict__ in,
                                             unsigned short* __restrict__ out, int n8){
  int idx = (int)blockIdx.x * 256 + TID;
  if (idx >= n8) return;
  ((uint4*)out)[idx] = cvt8(in + (size_t)idx * 8);
}

// ---------------- pack W_mem -> bf16 transposed [t][k] ----------------
__global__ __launch_bounds__(256) void k_pack(const float* __restrict__ Wmem,
                                              unsigned short* __restrict__ wmemT){
  int bh  = blockIdx.x;
  int idx = (int)blockIdx.y * 256 + TID;          // idx = t*DK + k
  int t = idx / DK, k = idx % DK;
  wmemT[(size_t)bh * (DT*DK) + idx] = f2bf(Wmem[(size_t)bh * (DK*DT) + (size_t)k*DT + t]);
}

// ---------------- pack mem rows (last 128 rows of hs per batch) -> bf16 ----------------
__global__ __launch_bounds__(256) void k_packmem(const float* __restrict__ hs,
                                                 unsigned short* __restrict__ memB){
  int idx = (int)blockIdx.x * 256 + TID;          // blocks of 8 halves
  if (idx >= Bn*NM*D/8) return;
  int flat = idx * 8;
  int b = flat / (NM*D), r = flat % (NM*D);
  ((uint4*)memB)[idx] = cvt8(hs + (size_t)b*S*D + (size_t)(S-NM)*D + r);
}

// ---------------- hs = from_heads(num/denom) + hidden ----------------
__global__ __launch_bounds__(256) void k_hs(const float* __restrict__ hidden,
                                            const unsigned short* __restrict__ hidB,
                                            const unsigned short* __restrict__ WqB,
                                            const unsigned short* __restrict__ wmemT,
                                            const float* __restrict__ z,
                                            float* __restrict__ hs){
  __shared__ __align__(16) unsigned short As[2][32*64];   // hidden tile bf16 (double buffer)
  __shared__ __align__(16) unsigned short Bs[2][64*64];   // Wq_h tile bf16  (double buffer)
  __shared__ __align__(16) unsigned short Mq[32*DK];      // normalized phi(q)
  __shared__ float Qs[32*64];
  __shared__ float r_ss[32], r_zd[32], r_den[32];

  const int stile = blockIdx.x, h = blockIdx.y, b = blockIdx.z;
  const int wave = TID >> 5, lane = TID & 31;
  const int hi8 = (lane >= 16) ? 8 : 0;
  const int srow0 = stile * 32;

  // ---- stage 1: q(32x64) = hidden_tile @ Wq_h^T, K=2048, chunk 64, double-buffered ----
  const int mi = wave >> 2, ni = wave & 3;
  const unsigned short* asrc = hidB + (size_t)(b*S + srow0 + (TID >> 3)) * D + (TID & 7) * 8;
  const unsigned short* bsrc0 = WqB + (size_t)(h*DH + (TID >> 3)) * D + (TID & 7) * 8;
  const unsigned short* bsrc1 = WqB + (size_t)(h*DH + ((TID+256) >> 3)) * D + (TID & 7) * 8;
  v8f accq = {};
  ((uint4*)As[0])[TID]       = *(const uint4*)(asrc);
  ((uint4*)Bs[0])[TID]       = *(const uint4*)(bsrc0);
  ((uint4*)Bs[0])[TID + 256] = *(const uint4*)(bsrc1);
  __syncthreads();
  for (int it = 0; it < 32; ++it){
    const int pp = it & 1;
    uint4 na, nb0, nb1;
    const bool more = (it + 1 < 32);
    if (more){
      int kn = (it + 1) * 64;
      na  = *(const uint4*)(asrc  + kn);
      nb0 = *(const uint4*)(bsrc0 + kn);
      nb1 = *(const uint4*)(bsrc1 + kn);
    }
#pragma unroll
    for (int kk = 0; kk < 64; kk += 32){
      bf16x16 a  = load_frag(As[pp], 64, mi*16, kk);
      bf16x16 bm = load_frag(Bs[pp], 64, ni*16, kk);
      accq = wmma_bf16(a, bm, accq);
    }
    if (more){
      ((uint4*)As[pp^1])[TID]       = na;
      ((uint4*)Bs[pp^1])[TID]       = nb0;
      ((uint4*)Bs[pp^1])[TID + 256] = nb1;
    }
    __syncthreads();
  }
#pragma unroll
  for (int r = 0; r < 8; ++r){
    int m = mi*16 + r + hi8, n = ni*16 + (lane & 15);
    Qs[m*64 + n] = accq[r];
  }
  if (TID < 32){ r_ss[TID] = 0.f; r_zd[TID] = 0.f; }
  __syncthreads();

  // ---- stage 2: phi, l2norm, denom = z.mq + eps ----
  const float* zh = z + (size_t)(b*H + h) * DK;
  const int row = TID >> 3, cbeg = (TID & 7) * 48;
  {
    const float* qr = Qs + row*64;
    float ss = 0.f, zd = 0.f;
    for (int c = cbeg; c < cbeg + 48; ++c){
      int n = c >> 7, p = c & 127, p2 = (p - (n+1)) & 127;
      float v = phix(qr, p) * phix(qr, p2);
      ss += v*v; zd += zh[c]*v;
    }
    atomicAdd(&r_ss[row], ss);
    atomicAdd(&r_zd[row], zd);
  }
  __syncthreads();
  {
    float inv = 1.f / fmaxf(sqrtf(r_ss[row]), 1e-12f);
    if ((TID & 7) == 0) r_den[row] = r_zd[row]*inv + EPS;
    const float* qr = Qs + row*64;
    for (int c = cbeg; c < cbeg + 48; ++c){
      int n = c >> 7, p = c & 127, p2 = (p - (n+1)) & 127;
      Mq[row*DK + c] = f2bf(phix(qr, p) * phix(qr, p2) * inv);
    }
  }
  __syncthreads();

  // ---- stage 3: num(32x256) = Mq @ W_mem_h ; out = num/den + residual ----
  const unsigned short* bt = wmemT + (size_t)(b*H + h) * (DT*DK);
  const int mi3 = wave & 1, nb = (wave >> 1) * 4;
  v8f acc[4] = {};
  for (int kc = 0; kc < DK; kc += 32){
    bf16x16 a = load_frag(Mq, DK, mi3*16, kc);
#pragma unroll
    for (int u = 0; u < 4; ++u){
      bf16x16 bm = load_frag(bt, DK, (nb+u)*16, kc);
      acc[u] = wmma_bf16(a, bm, acc[u]);
    }
  }
#pragma unroll
  for (int u = 0; u < 4; ++u)
#pragma unroll
    for (int r = 0; r < 8; ++r){
      int m = mi3*16 + r + hi8;
      int t = (nb+u)*16 + (lane & 15);
      size_t oi = (size_t)(b*S + srow0 + m) * D + h*DT + t;
      hs[oi] = acc[u][r] / r_den[m] + hidden[oi];
    }
}

// ---------------- mk, denom2, coef, z_new ----------------
__global__ __launch_bounds__(256) void k_mk(const unsigned short* __restrict__ memB,
                                            const unsigned short* __restrict__ WkB,
                                            const float* __restrict__ z,
                                            float* __restrict__ z_out,
                                            unsigned short* __restrict__ mkb,
                                            unsigned short* __restrict__ mkT,
                                            float* __restrict__ den_ws){
  __shared__ __align__(16) unsigned short As[2][NM*32];
  __shared__ __align__(16) unsigned short Bs[2][64*32];
  __shared__ float Qs[NM*64];
  __shared__ float r_ss[NM], r_zd[NM], r_inv[NM], r_cf[NM];

  const int bh = blockIdx.x, b = bh >> 3, h = bh & 7;
  const int wave = TID >> 5, lane = TID & 31;
  const int hi8 = (lane >= 16) ? 8 : 0;

  // q2(128x64) = mem @ Wk_h^T, K=2048, chunk 32, double-buffered
  // A: 512 uint4 (2/thread), B: 256 uint4 (1/thread)
  const unsigned short* a0 = memB + (size_t)(b*NM + (TID >> 2)) * D + (TID & 3) * 8;
  const unsigned short* a1 = memB + (size_t)(b*NM + ((TID+256) >> 2)) * D + (TID & 3) * 8;
  const unsigned short* b0 = WkB + (size_t)(h*DH + (TID >> 2)) * D + (TID & 3) * 8;
  v8f acc[4] = {};
  ((uint4*)As[0])[TID]       = *(const uint4*)a0;
  ((uint4*)As[0])[TID + 256] = *(const uint4*)a1;
  ((uint4*)Bs[0])[TID]       = *(const uint4*)b0;
  __syncthreads();
  for (int it = 0; it < 64; ++it){
    const int pp = it & 1;
    uint4 na0, na1, nb;
    const bool more = (it + 1 < 64);
    if (more){
      int kn = (it + 1) * 32;
      na0 = *(const uint4*)(a0 + kn);
      na1 = *(const uint4*)(a1 + kn);
      nb  = *(const uint4*)(b0 + kn);
    }
    bf16x16 a = load_frag(As[pp], 32, wave*16, 0);
#pragma unroll
    for (int u = 0; u < 4; ++u){
      bf16x16 bm = load_frag(Bs[pp], 32, u*16, 0);
      acc[u] = wmma_bf16(a, bm, acc[u]);
    }
    if (more){
      ((uint4*)As[pp^1])[TID]       = na0;
      ((uint4*)As[pp^1])[TID + 256] = na1;
      ((uint4*)Bs[pp^1])[TID]       = nb;
    }
    __syncthreads();
  }
#pragma unroll
  for (int u = 0; u < 4; ++u)
#pragma unroll
    for (int r = 0; r < 8; ++r){
      int m = wave*16 + r + hi8, n = u*16 + (lane & 15);
      Qs[m*64 + n] = acc[u][r];
    }
  if (TID < NM){ r_ss[TID] = 0.f; r_zd[TID] = 0.f; }
  __syncthreads();

  const float* zh = z + (size_t)bh * DK;
  const int row = TID >> 1, cbeg = (TID & 1) * 192;
  {
    const float* qr = Qs + row*64;
    float ss = 0.f, zd = 0.f;
    for (int c = cbeg; c < cbeg + 192; ++c){
      int n = c >> 7, p = c & 127, p2 = (p - (n+1)) & 127;
      float v = phix(qr, p) * phix(qr, p2);
      ss += v*v; zd += zh[c]*v;
    }
    atomicAdd(&r_ss[row], ss);
    atomicAdd(&r_zd[row], zd);
  }
  __syncthreads();
  if (TID < NM){
    float ss   = r_ss[TID];
    float inv  = 1.f / fmaxf(sqrtf(ss), 1e-12f);
    float den  = r_zd[TID]*inv + EPS;
    float mksq = ss*inv*inv;
    float cf   = fminf(fmaxf(1.f - den/mksq, 0.f), 1.f);
    r_inv[TID] = inv; r_cf[TID] = cf;
    den_ws[bh*NM + TID] = den;
  }
  __syncthreads();
  {
    const float* qr = Qs + row*64;
    float inv = r_inv[row];
    for (int c = cbeg; c < cbeg + 192; ++c){
      int n = c >> 7, p = c & 127, p2 = (p - (n+1)) & 127;
      unsigned short us = f2bf(phix(qr, p) * phix(qr, p2) * inv);
      mkb[(size_t)bh*(NM*DK) + (size_t)row*DK + c] = us;
      mkT[(size_t)bh*(NM*DK) + (size_t)c*NM + row] = us;
    }
  }
  for (int c = TID; c < DK; c += 256){
    int n = c >> 7, p = c & 127, p2 = (p - (n+1)) & 127;
    float a2 = z[(size_t)bh*DK + c];
    for (int j = 0; j < NM; ++j){
      const float* qr = Qs + j*64;
      a2 += r_cf[j] * phix(qr, p) * phix(qr, p2) * r_inv[j];
    }
    z_out[(size_t)bh*DK + c] = a2;
  }
}

// ---------------- mvb = (new_mv - num2/denom2) * mb ----------------
__global__ __launch_bounds__(256) void k_mv(const float* __restrict__ hs,
                                            const unsigned short* __restrict__ memB,
                                            const unsigned short* __restrict__ WvB,
                                            const float* __restrict__ Wbw,
                                            const float* __restrict__ Wbb,
                                            const unsigned short* __restrict__ mkb,
                                            const unsigned short* __restrict__ wmemT,
                                            const float* __restrict__ den_ws,
                                            unsigned short* __restrict__ mvbT){
  __shared__ __align__(16) unsigned short As[2][NM*32];
  __shared__ __align__(16) unsigned short Bs[2][NM*32];
  __shared__ float part[256];
  __shared__ float r_mb[NM], r_den[NM];

  const int bh = blockIdx.x, b = bh >> 3, h = bh & 7;
  const int nh = blockIdx.y;
  const int wave = TID >> 5, lane = TID & 31;
  const int hi8 = (lane >= 16) ? 8 : 0;
  const int mrow0 = b*S + (S - NM);

  // mb[j] = sigmoid(mem[j] . Wbw_h + bb)
  {
    int j  = TID >> 1;
    int k0 = (TID & 1) * 1024;
    const float* mr = hs  + (size_t)(mrow0 + j) * D;
    const float* wr = Wbw + (size_t)h * D;
    float s = 0.f;
    for (int k = k0; k < k0 + 1024; ++k) s += mr[k] * wr[k];
    part[TID] = s;
  }
  __syncthreads();
  if (TID < NM){
    float x = part[2*TID] + part[2*TID+1] + Wbb[h];
    r_mb[TID]  = 1.f / (1.f + __expf(-x));
    r_den[TID] = den_ws[bh*NM + TID];
  }
  __syncthreads();

  // new_mv(128x128) = mem @ Wv_h^T (half nh), K=2048, chunk 32, double-buffered
  const unsigned short* a0 = memB + (size_t)(b*NM + (TID >> 2)) * D + (TID & 3) * 8;
  const unsigned short* a1 = memB + (size_t)(b*NM + ((TID+256) >> 2)) * D + (TID & 3) * 8;
  const unsigned short* b0 = WvB + (size_t)(h*DT + nh*NM + (TID >> 2)) * D + (TID & 3) * 8;
  const unsigned short* b1 = WvB + (size_t)(h*DT + nh*NM + ((TID+256) >> 2)) * D + (TID & 3) * 8;
  v8f cmv[8] = {};
  ((uint4*)As[0])[TID]       = *(const uint4*)a0;
  ((uint4*)As[0])[TID + 256] = *(const uint4*)a1;
  ((uint4*)Bs[0])[TID]       = *(const uint4*)b0;
  ((uint4*)Bs[0])[TID + 256] = *(const uint4*)b1;
  __syncthreads();
  for (int it = 0; it < 64; ++it){
    const int pp = it & 1;
    uint4 na0, na1, nb0, nb1;
    const bool more = (it + 1 < 64);
    if (more){
      int kn = (it + 1) * 32;
      na0 = *(const uint4*)(a0 + kn);
      na1 = *(const uint4*)(a1 + kn);
      nb0 = *(const uint4*)(b0 + kn);
      nb1 = *(const uint4*)(b1 + kn);
    }
    bf16x16 a = load_frag(As[pp], 32, wave*16, 0);
#pragma unroll
    for (int u = 0; u < 8; ++u){
      bf16x16 bm = load_frag(Bs[pp], 32, u*16, 0);
      cmv[u] = wmma_bf16(a, bm, cmv[u]);
    }
    if (more){
      ((uint4*)As[pp^1])[TID]       = na0;
      ((uint4*)As[pp^1])[TID + 256] = na1;
      ((uint4*)Bs[pp^1])[TID]       = nb0;
      ((uint4*)Bs[pp^1])[TID + 256] = nb1;
    }
    __syncthreads();
  }

  // num2(128x128) = mk @ W_mem_h (half nh), K=384, operands direct from L2
  v8f cn2[8] = {};
  const unsigned short* am  = mkb   + (size_t)bh * (NM*DK);
  const unsigned short* bm2 = wmemT + (size_t)bh * (DT*DK);
  for (int kc = 0; kc < DK; kc += 32){
    bf16x16 a = load_frag(am, DK, wave*16, kc);
#pragma unroll
    for (int u = 0; u < 8; ++u){
      bf16x16 bfr = load_frag(bm2, DK, nh*NM + u*16, kc);
      cn2[u] = wmma_bf16(a, bfr, cn2[u]);
    }
  }

#pragma unroll
  for (int u = 0; u < 8; ++u)
#pragma unroll
    for (int r = 0; r < 8; ++r){
      int j = wave*16 + r + hi8;
      int t = nh*NM + u*16 + (lane & 15);
      float val = (cmv[u][r] - cn2[u][r] / r_den[j]) * r_mb[j];
      mvbT[(size_t)bh*(DT*NM) + (size_t)t*NM + j] = f2bf(val);
    }
}

// ---------------- W_mem_new = W_mem + mk^T @ mvb ----------------
__global__ __launch_bounds__(256) void k_assoc(const unsigned short* __restrict__ mkT,
                                               const unsigned short* __restrict__ mvbT,
                                               const float* __restrict__ Wmem,
                                               float* __restrict__ Wout){
  const int bh = blockIdx.x;
  const int kbase = (int)blockIdx.y * 64;
  const int wave = TID >> 5, lane = TID & 31;
  const int hi8 = (lane >= 16) ? 8 : 0;
  const int mi = wave >> 1, nb = (wave & 1) * 8;
  const unsigned short* a0 = mkT  + (size_t)bh * (DK*NM);
  const unsigned short* b0 = mvbT + (size_t)bh * (DT*NM);
  v8f acc[8] = {};
  for (int kc = 0; kc < NM; kc += 32){
    bf16x16 a = load_frag(a0, NM, kbase + mi*16, kc);
#pragma unroll
    for (int u = 0; u < 8; ++u){
      bf16x16 bm = load_frag(b0, NM, (nb+u)*16, kc);
      acc[u] = wmma_bf16(a, bm, acc[u]);
    }
  }
#pragma unroll
  for (int u = 0; u < 8; ++u)
#pragma unroll
    for (int r = 0; r < 8; ++r){
      int k = kbase + mi*16 + r + hi8;
      int t = (nb+u)*16 + (lane & 15);
      size_t oi = (size_t)bh*(DK*DT) + (size_t)k*DT + t;
      Wout[oi] = Wmem[oi] + acc[u][r];
    }
}

extern "C" void kernel_launch(void* const* d_in, const int* in_sizes, int n_in,
                              void* d_out, int out_size, void* d_ws, size_t ws_size,
                              hipStream_t stream){
  const float* hidden = (const float*)d_in[0];
  const float* Wmem   = (const float*)d_in[1];
  const float* z      = (const float*)d_in[2];
  const float* Wq     = (const float*)d_in[3];
  const float* Wk     = (const float*)d_in[4];
  const float* Wv     = (const float*)d_in[5];
  const float* Wbw    = (const float*)d_in[6];
  const float* Wbb    = (const float*)d_in[7];

  float* hs   = (float*)d_out;                          // (4,2048,2048)
  float* Wout = hs   + (size_t)Bn*S*D;                  // (4,8,384,256)
  float* zout = Wout + (size_t)Bn*H*DK*DT;              // (4,8,384)

  char* ws = (char*)d_ws;
  unsigned short* wmemT = (unsigned short*)ws; ws += (size_t)Bn*H*DT*DK*2;  // 6.0 MB
  unsigned short* mkb   = (unsigned short*)ws; ws += (size_t)Bn*H*NM*DK*2;  // 3.0 MB
  unsigned short* mkTr  = (unsigned short*)ws; ws += (size_t)Bn*H*NM*DK*2;  // 3.0 MB
  unsigned short* mvbT  = (unsigned short*)ws; ws += (size_t)Bn*H*DT*NM*2;  // 2.0 MB
  unsigned short* hidB  = (unsigned short*)ws; ws += (size_t)Bn*S*D*2;      // 33.5 MB
  unsigned short* WqB   = (unsigned short*)ws; ws += (size_t)(H*DH)*D*2;    // 2.0 MB
  unsigned short* WkB   = (unsigned short*)ws; ws += (size_t)(H*DH)*D*2;    // 2.0 MB
  unsigned short* WvB   = (unsigned short*)ws; ws += (size_t)D*D*2;         // 8.0 MB
  unsigned short* memB  = (unsigned short*)ws; ws += (size_t)Bn*NM*D*2;     // 2.0 MB
  float*          denw  = (float*)ws;                                       // 16 KB

  const int nHid8 = Bn*S*D/8, nWq8 = (H*DH)*D/8, nWv8 = D*D/8, nMem8 = Bn*NM*D/8;
  k_cvt  <<<dim3((nHid8+255)/256), 256, 0, stream>>>(hidden, hidB, nHid8);
  k_cvt  <<<dim3((nWq8 +255)/256), 256, 0, stream>>>(Wq, WqB, nWq8);
  k_cvt  <<<dim3((nWq8 +255)/256), 256, 0, stream>>>(Wk, WkB, nWq8);
  k_cvt  <<<dim3((nWv8 +255)/256), 256, 0, stream>>>(Wv, WvB, nWv8);
  k_pack <<<dim3(Bn*H, (DT*DK)/256), 256, 0, stream>>>(Wmem, wmemT);
  k_hs   <<<dim3(S/32, H, Bn),       256, 0, stream>>>(hidden, hidB, WqB, wmemT, z, hs);
  k_packmem<<<dim3((nMem8+255)/256), 256, 0, stream>>>(hs, memB);
  k_mk   <<<dim3(Bn*H),              256, 0, stream>>>(memB, WkB, z, zout, mkb, mkTr, denw);
  k_mv   <<<dim3(Bn*H, 2),           256, 0, stream>>>(hs, memB, WvB, Wbw, Wbb, mkb, wmemT, denw, mvbT);
  k_assoc<<<dim3(Bn*H, DK/64),       256, 0, stream>>>(mkTr, mvbT, Wmem, Wout);
}
